// GAC_encoder_80917183857397
// MI455X (gfx1250) — matrix operand order repaired and checked by reading, not compiled, not run on hardware
//
#include <hip/hip_runtime.h>

typedef __attribute__((ext_vector_type(16))) _Float16 v16h;
typedef __attribute__((ext_vector_type(8)))  float    v8f;

#define NP  12000
#define KNB 24

// ======================= KNN: top-24 nearest per point =======================
// Block = 128 query points. Candidates streamed through LDS in tiles of 32.
// Per-thread sorted top-24 kept in registers (fully unrolled bubble insert).
template<int CD>
__global__ __launch_bounds__(128)
void knn_kernel(const float* __restrict__ C, int* __restrict__ idxOut) {
  __shared__ float sq[CD * 128];   // [ch][query]  (lane-consecutive -> no bank conflict)
  __shared__ float sc[CD * 32];    // [ch][cand]   (broadcast reads)
  const int tid = threadIdx.x;
  const int qg  = blockIdx.x * 128 + tid;

  for (int i = tid; i < CD * 128; i += 128) {
    int ql = i & 127, ch = i >> 7;
    int gq = blockIdx.x * 128 + ql;
    sq[i] = (gq < NP) ? C[(long)ch * NP + gq] : 0.f;
  }

  float best[KNB]; int bidx[KNB];
#pragma unroll
  for (int j = 0; j < KNB; ++j) { best[j] = 3.4e38f; bidx[j] = 0; }

  for (int tile = 0; tile < NP; tile += 32) {   // 12000 % 32 == 0
    __syncthreads();
    for (int i = tid; i < CD * 32; i += 128) {
      int cc = i & 31, ch = i >> 5;
      sc[i] = C[(long)ch * NP + tile + cc];
    }
    __syncthreads();
    for (int cc = 0; cc < 32; ++cc) {
      float d2 = 0.f;
#pragma unroll 8
      for (int ch = 0; ch < CD; ++ch) {
        float df = sq[ch * 128 + tid] - sc[ch * 32 + cc];
        d2 = fmaf(df, df, d2);
      }
      if (d2 < best[KNB - 1]) {
        float cd = d2; int ci = tile + cc;
#pragma unroll
        for (int j = 0; j < KNB; ++j) {
          if (cd < best[j]) {
            float tf = best[j]; best[j] = cd; cd = tf;
            int ti = bidx[j];  bidx[j] = ci; ci = ti;
          }
        }
      }
    }
  }
  if (qg < NP) {
#pragma unroll
    for (int j = 0; j < KNB; ++j) idxOut[qg * KNB + j] = bidx[j];
  }
}

// ======================= weight pre-convert f32 -> f16 (padded pitch) ========
__global__ __launch_bounds__(256)
void wcvt_kernel(const float* __restrict__ W, _Float16* __restrict__ W16,
                 int Co, int Cin, int pitch) {
  int i = blockIdx.x * 256 + threadIdx.x;
  if (i < Co * pitch) {
    int o = i / pitch, c = i - o * pitch;
    W16[i] = (_Float16)((c < Cin) ? W[(long)o * Cin + c] : 0.f);
  }
}

// ======================= lsam conv (16x9) + BN + LReLU, and distance d =======
// One thread per flat column gid = a*12000 + n2  (24*12000 total).
__global__ __launch_bounds__(256)
void lsam_kernel(const float* __restrict__ X, int co,
                 const int* __restrict__ idx,
                 const float* __restrict__ lw, const float* __restrict__ lb,
                 const float* __restrict__ g,  const float* __restrict__ b2,
                 const float* __restrict__ m2, const float* __restrict__ v2,
                 float* __restrict__ rbuf, float* __restrict__ dbuf) {
  int gid = blockIdx.x * 256 + threadIdx.x;        // exactly 1125*256 = 288000
  int a = gid / NP, n2 = gid - a * NP;
  int k = n2 % KNB, q = n2 / KNB;

  float acc[16];
#pragma unroll
  for (int o = 0; o < 16; ++o) acc[o] = 0.f;

#pragma unroll
  for (int c = 0; c < 9; ++c) {
    int t  = a * 9 + c;
    int u  = t / KNB;                               // lsam channel 0..8
    int n  = (t - u * KNB) * (NP / KNB) + q;        // source point
    int gr = u / 3, cc = co + (u - gr * 3);         // coord channel
    float val;
    if (gr == 0)      { int m = idx[n * KNB + k]; val = X[(long)cc * NP + m] - X[(long)cc * NP + n]; }
    else if (gr == 1) { val = X[(long)cc * NP + n]; }
    else              { int m = idx[n * KNB + k]; val = X[(long)cc * NP + m]; }
#pragma unroll
    for (int o = 0; o < 16; ++o) acc[o] = fmaf(lw[o * 9 + c], val, acc[o]);
  }
#pragma unroll
  for (int o = 0; o < 16; ++o) {
    float s = g[o] * rsqrtf(v2[o] + 1e-5f);
    float v = (acc[o] + lb[o] - m2[o]) * s + b2[o];
    v = (v >= 0.f) ? v : 0.01f * v;
    rbuf[((long)(a * 16 + o)) * NP + n2] = v;
  }
  // d = sqrt(sum((cj - ci + 1e-6)^2)); flat layout [a][n2] == n*24+k
  int nd = gid / KNB, kd = gid - nd * KNB;
  int m  = idx[nd * KNB + kd];
  float s2 = 0.f;
#pragma unroll
  for (int cc = 0; cc < 3; ++cc) {
    float df = X[(long)(co + cc) * NP + m] - X[(long)(co + cc) * NP + nd] + 1e-6f;
    s2 = fmaf(df, df, s2);
  }
  dbuf[gid] = sqrtf(s2);
}

// ======================= fused gather for GEMM B operand =====================
// MODE 0: f-conv input  (channels: [0,2F)=z gather, [2F,2F+16)=r buffer)
// MODE 1: attn input    ([0,F)=feat-ctr, [F,2F)=feat, 2F=d)
// MODE 2: dense matrix read (final conv)
template<int MODE>
__device__ inline float gatherB(const float* __restrict__ X, const int* __restrict__ idx,
                                const float* __restrict__ rbuf, const float* __restrict__ dbuf,
                                int col, int c, int F, int Cin) {
  if (MODE == 2) return (c < Cin) ? X[(long)c * NP + col] : 0.f;
  int a = col / NP, n2 = col - a * NP;
  int k = n2 % KNB, q = n2 / KNB;
  if (MODE == 0) {
    if (c < 2 * F) {
      int t = a * 2 * F + c;
      int u = t / KNB;
      int n = (t - u * KNB) * (NP / KNB) + q;
      if (u < F) return X[(long)u * NP + n];               // group 1: ctr
      int m = idx[n * KNB + k];
      return X[(long)(u - F) * NP + m];                    // group 2: feat
    } else if (c < 2 * F + 16) {
      return rbuf[((long)(a * 16 + (c - 2 * F))) * NP + n2];
    }
    return 0.f;
  } else {
    if (c < F) {
      int t = a * F + c, u = t / KNB;
      int n = (t - u * KNB) * (NP / KNB) + q;
      int m = idx[n * KNB + k];
      return X[(long)u * NP + m] - X[(long)u * NP + n];    // group 0: feat - ctr
    } else if (c < 2 * F) {
      int t = a * F + (c - F), u = t / KNB;
      int n = (t - u * KNB) * (NP / KNB) + q;
      int m = idx[n * KNB + k];
      return X[(long)u * NP + m];                          // group 2: feat
    } else if (c == 2 * F) {
      return dbuf[(long)a * NP + n2];
    }
    return 0.f;
  }
}

// ======================= WMMA GEMM: out = act(bn(W @ gathered + b)) ==========
// 256 threads = 8 waves, 128 columns per workgroup (16 per wave), all Co rows
// per wave (RT = Co/16 accumulators). K-chunks of 32 staged in LDS as f16.
// Weights come pre-converted/padded (pitch % 32 == 0), so A-staging is a pure
// guard-free uint4 copy (no exec churn, no per-element load->cvt wait chains).
// LDS row stride 40 halfs (80B): 16B-aligned fragment sub-reads + bank spread.
template<int MODE, int RT>
__global__ __launch_bounds__(256)
void gemm_kernel(const float* __restrict__ X, const int* __restrict__ idx,
                 const float* __restrict__ rbuf, const float* __restrict__ dbuf,
                 const _Float16* __restrict__ W16, const float* __restrict__ cb,
                 const float* __restrict__ g,  const float* __restrict__ bnb,
                 const float* __restrict__ bnm, const float* __restrict__ bnv,
                 int F, int Cin, int pitch, int Co, long NC,
                 float* __restrict__ out) {
  __shared__ __attribute__((aligned(16))) _Float16 ldsA[256 * 40];
  __shared__ __attribute__((aligned(16))) _Float16 ldsB[128 * 40];
  __shared__ float ldsAl[256], ldsBe[256];

  const int tid  = threadIdx.x;
  const int wave = tid >> 5;
  const int lane = tid & 31;
  const int hh   = lane >> 4;        // wave half
  const int ln   = lane & 15;
  const long colBase = (long)blockIdx.x * 128;

  for (int o = tid; o < Co; o += 256) {
    if (MODE == 2) { ldsAl[o] = 1.f; ldsBe[o] = cb[o]; }
    else {
      float s = g[o] * rsqrtf(bnv[o] + 1e-5f);
      ldsAl[o] = s;
      ldsBe[o] = (cb[o] - bnm[o]) * s + bnb[o];
    }
  }

  v8f acc[RT];
#pragma unroll
  for (int rt = 0; rt < RT; ++rt)
#pragma unroll
    for (int e = 0; e < 8; ++e) acc[rt][e] = 0.f;

  for (int kb = 0; kb < pitch; kb += 32) {
    __syncthreads();
    // Stage A (pre-converted f16 weights): 16B per thread per pass, no guards.
    // Co*4 is a multiple of 256 for Co in {64,128,256}.
    for (int i = tid; i < Co * 4; i += 256) {
      int o = i >> 2, seg = i & 3;
      *(uint4*)(ldsA + o * 40 + seg * 8) =
          *(const uint4*)(W16 + (long)o * pitch + kb + seg * 8);
    }
    if (kb + 32 < pitch && tid < Co)
      __builtin_prefetch(W16 + (long)tid * pitch + kb + 32, 0, 1);
    // Stage B (gathered input columns), 128*32 elements; unroll for MLP.
#pragma unroll 4
    for (int i = tid; i < 128 * 32; i += 256) {
      int cl = i >> 5, kk = i & 31, c = kb + kk;
      long col = colBase + cl;
      float v = 0.f;
      if (col < NC) v = gatherB<MODE>(X, idx, rbuf, dbuf, (int)col, c, F, Cin);
      ldsB[cl * 40 + kk] = (_Float16)v;
    }
    __syncthreads();

    // B fragment (32x16): lane(n=ln, half=hh), element e -> K = e + 16*hh
    v16h bfrag;
    {
      int cl = wave * 16 + ln;
#pragma unroll
      for (int e = 0; e < 16; ++e) bfrag[e] = ldsB[cl * 40 + 16 * hh + e];
    }
#pragma unroll
    for (int rt = 0; rt < RT; ++rt) {
      // A fragment (16x32): row m = ln; element e -> K = e + 16*hh
      v16h afrag;
      int m = rt * 16 + ln;
#pragma unroll
      for (int e = 0; e < 16; ++e) {
        int kk = e + 16 * hh;
        afrag[e] = ldsA[m * 40 + kk];
      }
      acc[rt] = __builtin_amdgcn_wmma_f32_16x16x32_f16(
          false, afrag, false, bfrag, (short)0, acc[rt], false, false);
    }
  }

  // Epilogue: C/D layout lane -> col n=ln, VGPR v -> row v + 8*hh
  long col = colBase + wave * 16 + ln;
  bool wok = col < NC;
  int a  = (int)(col / NP);
  int n2 = (int)(col - (long)a * NP);
#pragma unroll
  for (int rt = 0; rt < RT; ++rt)
#pragma unroll
    for (int vv = 0; vv < 8; ++vv) {
      int o = rt * 16 + vv + 8 * hh;
      float val = acc[rt][vv] * ldsAl[o] + ldsBe[o];
      if (MODE != 2) val = (val >= 0.f) ? val : 0.01f * val;
      if (wok) {
        if (MODE == 2) out[(long)o * NP + n2] = val;
        else           out[((long)(a * Co + o)) * NP + n2] = val;
      }
    }
}

// ======================= softmax over channels + sum over K ==================
// out[o][n2] = sum_a softmax_o(attn[a][:][n2])[o] * f[a][o][n2]
__global__ __launch_bounds__(64)
void softmax_reduce(const float* __restrict__ attn, const float* __restrict__ fcap,
                    int Co, float* __restrict__ dst0, float* __restrict__ dst1) {
  extern __shared__ float accS[];          // [Co][64]
  const int tid = threadIdx.x;
  const int n2  = blockIdx.x * 64 + tid;
  for (int o = 0; o < Co; ++o) accS[o * 64 + tid] = 0.f;
  if (n2 < NP) {
    for (int a = 0; a < KNB; ++a) {
      long base = ((long)a * Co) * NP + n2;
      float mx = -3.4e38f;
      for (int o = 0; o < Co; ++o) mx = fmaxf(mx, attn[base + (long)o * NP]);
      float sum = 0.f;
      for (int o = 0; o < Co; ++o) sum += __expf(attn[base + (long)o * NP] - mx);
      float inv = 1.f / sum;
      for (int o = 0; o < Co; ++o) {
        float w = __expf(attn[base + (long)o * NP] - mx) * inv;
        accS[o * 64 + tid] += w * fcap[base + (long)o * NP];
      }
    }
    for (int o = 0; o < Co; ++o) {
      float v = accS[o * 64 + tid];
      dst0[(long)o * NP + n2] = v;
      if (dst1) dst1[(long)o * NP + n2] = v;
    }
  }
}

// ======================= append xyz rows =====================================
__global__ void copy3(const float* __restrict__ s, float* __restrict__ d) {
  int i = blockIdx.x * 256 + threadIdx.x;
  if (i < 3 * NP) d[i] = s[i];
}

// ======================= host launcher =======================================
extern "C" void kernel_launch(void* const* d_in, const int* in_sizes, int n_in,
                              void* d_out, int out_size, void* d_ws, size_t ws_size,
                              hipStream_t stream) {
  (void)in_sizes; (void)n_in; (void)out_size; (void)ws_size;
  const float* x = (const float*)d_in[0];
  const float* P[57];
  for (int i = 0; i < 57; ++i) P[i] = (const float*)d_in[i];

  // workspace carve (all offsets 128B aligned); total ~613 MB
  char* ws = (char*)d_ws;
  int*   idxb = (int*)(ws);                                   //  288000 i32
  float* rbuf = (float*)(ws + 1152000);                        //  24*16*12000 f32
  float* dbuf = (float*)(ws + 1152000 + 18432000);             //  288000 f32
  float* fpre = (float*)(ws + 20736000);                       //  24*256*12000 f32
  float* apre = (float*)(ws + 20736000 + 294912000);           //  24*256*12000 f32
  float* res1 = (float*)(ws + 610560000);                      //  67*12000 f32
  float* res2 = (float*)(ws + 613776000);                      // 131*12000 f32
  float* cat  = (float*)(ws + 620064000);                      // 448*12000 f32
  // f16 weight pool (padded pitches)
  _Float16* w1f = (_Float16*)(ws + 641568000);                 //  64 x  64
  _Float16* w1a = (_Float16*)(ws + 641568000 +   8192);        //  64 x  64
  _Float16* w2f = (_Float16*)(ws + 641568000 +  16384);        // 128 x 160
  _Float16* w2a = (_Float16*)(ws + 641568000 +  57344);        // 128 x 160
  _Float16* w3f = (_Float16*)(ws + 641568000 +  98304);        // 256 x 288
  _Float16* w3a = (_Float16*)(ws + 641568000 + 245760);        // 256 x 288
  _Float16* wfn = (_Float16*)(ws + 641568000 + 393216);        // 256 x 448

  // -------- pre-convert weights to f16 (tiny, one-time per launch) ---------
  wcvt_kernel<<< 16, 256, 0, stream>>>(P[7],  w1f,  64,  64,  64);
  wcvt_kernel<<< 16, 256, 0, stream>>>(P[13], w1a,  64,  49,  64);
  wcvt_kernel<<< 80, 256, 0, stream>>>(P[25], w2f, 128, 144, 160);
  wcvt_kernel<<< 80, 256, 0, stream>>>(P[31], w2a, 128, 129, 160);
  wcvt_kernel<<<288, 256, 0, stream>>>(P[43], w3f, 256, 272, 288);
  wcvt_kernel<<<288, 256, 0, stream>>>(P[49], w3a, 256, 257, 288);
  wcvt_kernel<<<448, 256, 0, stream>>>(P[55], wfn, 256, 448, 448);

  // ---------------- Block 1 : D=24, F=24, co=9, Co=64 ----------------
  knn_kernel<3><<<94, 128, 0, stream>>>(x + 9 * NP, idxb);
  lsam_kernel<<<1125, 256, 0, stream>>>(x, 9, idxb,
      P[1], P[2], P[3], P[4], P[5], P[6], rbuf, dbuf);
  gemm_kernel<0, 4><<<2250, 256, 0, stream>>>(x, idxb, rbuf, dbuf,
      w1f, P[8], P[9], P[10], P[11], P[12], 24, 64, 64, 64, 288000L, fpre);
  gemm_kernel<1, 4><<<2250, 256, 0, stream>>>(x, idxb, rbuf, dbuf,
      w1a, P[14], P[15], P[16], P[17], P[18], 24, 49, 64, 64, 288000L, apre);
  softmax_reduce<<<188, 64, 64 * 64 * sizeof(float), stream>>>(apre, fpre, 64, res1, cat);
  copy3<<<141, 256, 0, stream>>>(x + 9 * NP, res1 + 64 * NP);

  // ---------------- Block 2 : D=67, F=64, co=64, Co=128 --------------
  knn_kernel<64><<<94, 128, 0, stream>>>(res1, idxb);
  lsam_kernel<<<1125, 256, 0, stream>>>(res1, 64, idxb,
      P[19], P[20], P[21], P[22], P[23], P[24], rbuf, dbuf);
  gemm_kernel<0, 8><<<2250, 256, 0, stream>>>(res1, idxb, rbuf, dbuf,
      w2f, P[26], P[27], P[28], P[29], P[30], 64, 144, 160, 128, 288000L, fpre);
  gemm_kernel<1, 8><<<2250, 256, 0, stream>>>(res1, idxb, rbuf, dbuf,
      w2a, P[32], P[33], P[34], P[35], P[36], 64, 129, 160, 128, 288000L, apre);
  softmax_reduce<<<188, 64, 64 * 128 * sizeof(float), stream>>>(apre, fpre, 128, res2, cat + 64 * NP);
  copy3<<<141, 256, 0, stream>>>(x + 9 * NP, res2 + 128 * NP);

  // ---------------- Block 3 : D=131, F=128, co=128, Co=256 -----------
  knn_kernel<128><<<94, 128, 0, stream>>>(res2, idxb);
  lsam_kernel<<<1125, 256, 0, stream>>>(res2, 128, idxb,
      P[37], P[38], P[39], P[40], P[41], P[42], rbuf, dbuf);
  gemm_kernel<0, 16><<<2250, 256, 0, stream>>>(res2, idxb, rbuf, dbuf,
      w3f, P[44], P[45], P[46], P[47], P[48], 128, 272, 288, 256, 288000L, fpre);
  gemm_kernel<1, 16><<<2250, 256, 0, stream>>>(res2, idxb, rbuf, dbuf,
      w3a, P[50], P[51], P[52], P[53], P[54], 128, 257, 288, 256, 288000L, apre);
  softmax_reduce<<<188, 64, 64 * 256 * sizeof(float), stream>>>(apre, fpre, 256, cat + 192 * NP, nullptr);

  // ---------------- Final conv 256x448, no BN/activation -------------
  gemm_kernel<2, 16><<<94, 256, 0, stream>>>(cat, nullptr, nullptr, nullptr,
      wfn, P[56], nullptr, nullptr, nullptr, nullptr, 0, 448, 448, 256, (long)NP, (float*)d_out);
}
